// DinoDecoderBlock_65317862637919
// MI455X (gfx1250) — compile-verified, matrix operand order checked by reading, not run on hardware
//
#include <hip/hip_runtime.h>
#include <hip/hip_bf16.h>

typedef _Float16 half_t;
typedef __attribute__((ext_vector_type(16))) _Float16 v16h;
typedef __attribute__((ext_vector_type(8)))  float    v8f;

#define BB 4
#define NQ 1024
#define NKK 1024
#define CC 768
#define HH 12
#define DHH 64
#define MM (BB*NQ)          // 4096 rows
static __device__ __constant__ float kScale = 0.125f;  // 64^-0.5

// ---------------------------------------------------------------------------
// f32 -> f16 conversion (weights)
// ---------------------------------------------------------------------------
__global__ void cvt_f16_kernel(const float* __restrict__ in, half_t* __restrict__ out, int n) {
    int i = blockIdx.x * blockDim.x + threadIdx.x;
    if (i < n) out[i] = (half_t)in[i];
}

// ---------------------------------------------------------------------------
// mean over heads of similarities: out[b,q,k] = mean_h sims[b,h,q,k]
// ---------------------------------------------------------------------------
__global__ void head_mean_kernel(const float* __restrict__ sims, float* __restrict__ out) {
    int i = blockIdx.x * blockDim.x + threadIdx.x;
    int n = BB * NQ * NKK;
    if (i >= n) return;
    int b  = i / (NQ * NKK);
    int qk = i - b * (NQ * NKK);
    const float* p = sims + (size_t)b * HH * NQ * NKK + qk;
    float s = 0.f;
#pragma unroll
    for (int h = 0; h < HH; ++h) s += p[(size_t)h * NQ * NKK];
    out[i] = s * (1.0f / HH);
}

// ---------------------------------------------------------------------------
// LayerNorm: one 256-thread block per row (8 waves), f32 in -> f16 out
// ---------------------------------------------------------------------------
__global__ __launch_bounds__(256)
void layernorm_f16_kernel(const float* __restrict__ x, const float* __restrict__ g,
                          const float* __restrict__ b, half_t* __restrict__ out) {
    __shared__ float red[16];
    int row = blockIdx.x;
    const float* xr = x + (size_t)row * CC;
    float s = 0.f, s2 = 0.f;
    for (int i = threadIdx.x; i < CC; i += 256) { float v = xr[i]; s += v; s2 += v * v; }
#pragma unroll
    for (int o = 16; o > 0; o >>= 1) { s += __shfl_xor(s, o); s2 += __shfl_xor(s2, o); }
    int wave = threadIdx.x >> 5, lane = threadIdx.x & 31;
    if (lane == 0) { red[wave * 2] = s; red[wave * 2 + 1] = s2; }
    __syncthreads();
    float ts = 0.f, ts2 = 0.f;
#pragma unroll
    for (int w = 0; w < 8; ++w) { ts += red[w * 2]; ts2 += red[w * 2 + 1]; }
    float mean = ts * (1.0f / CC);
    float var  = ts2 * (1.0f / CC) - mean * mean;
    float r    = rsqrtf(var + 1e-5f);
    half_t* orow = out + (size_t)row * CC;
    for (int i = threadIdx.x; i < CC; i += 256)
        orow[i] = (half_t)((xr[i] - mean) * r * g[i] + b[i]);
}

// ---------------------------------------------------------------------------
// V transpose: vt[(b*H+h)*DH + d][k] = v[(b*NK + k)*rowStride + h*DH + d]
// (source pointer pre-offset to the right column block, e.g. +2C for qkv's V)
// ---------------------------------------------------------------------------
__global__ void transpose_v_kernel(const half_t* __restrict__ v, half_t* __restrict__ vt,
                                   int rowStride) {
    int bh = blockIdx.z, b = bh / HH;
    int d  = blockIdx.y;
    int k  = blockIdx.x * 64 + threadIdx.x;
    int h  = bh - b * HH;
    vt[((size_t)bh * DHH + d) * NKK + k] =
        v[(size_t)(b * NKK + k) * rowStride + h * DHH + d];
}

// ---------------------------------------------------------------------------
// WMMA GEMM: D[M,N] = act( A[M,K](f16) @ W[N,K](f16)^T + bias + residual )
// Block tile 128M x 64N, 8 waves (4x2), each wave 32x32 (2x2 wmma tiles),
// k-step 32, LDS double buffered.
// ---------------------------------------------------------------------------
template<bool HAS_BIAS, bool RESIDUAL, bool GELU_OUT, bool OUT_F16>
__global__ __launch_bounds__(256)
void gemm_wmma_kernel(const half_t* __restrict__ A, const half_t* __restrict__ W,
                      const float* __restrict__ bias, const float* __restrict__ res,
                      half_t* __restrict__ outH, float* __restrict__ outF,
                      int M, int N, int K) {
    __shared__ __align__(32) half_t lsA[2][128 * 32];
    __shared__ __align__(32) half_t lsW[2][64 * 32];
    const int tid  = threadIdx.x;
    const int wave = tid >> 5, lane = tid & 31;
    const int wm = wave >> 1, wn = wave & 1;       // 4 x 2 wave grid
    const int l2 = lane >> 4, l16 = lane & 15;
    const int mblk = blockIdx.x * 128;
    const int nblk = blockIdx.y * 64;

    const int arow = tid >> 1, acol = (tid & 1) * 16;   // 16 halves / thread
    const int wrow = tid >> 2, wcol = (tid & 3) * 8;    // 8 halves / thread

    v8f c[2][2] = {};
    const int nsteps = K >> 5;

    { // prologue: stage 0
        const half_t* ga = A + (size_t)(mblk + arow) * K + acol;
        *(int4*)&lsA[0][arow * 32 + acol]     = *(const int4*)ga;
        *(int4*)&lsA[0][arow * 32 + acol + 8] = *(const int4*)(ga + 8);
        const half_t* gw = W + (size_t)(nblk + wrow) * K + wcol;
        *(int4*)&lsW[0][wrow * 32 + wcol] = *(const int4*)gw;
    }
    __syncthreads();

    for (int ks = 0; ks < nsteps; ++ks) {
        const int cur = ks & 1, nxt = cur ^ 1;
        if (ks + 1 < nsteps) {
            const int kofs = (ks + 1) << 5;
            const half_t* ga = A + (size_t)(mblk + arow) * K + kofs + acol;
            *(int4*)&lsA[nxt][arow * 32 + acol]     = *(const int4*)ga;
            *(int4*)&lsA[nxt][arow * 32 + acol + 8] = *(const int4*)(ga + 8);
            const half_t* gw = W + (size_t)(nblk + wrow) * K + kofs + wcol;
            *(int4*)&lsW[nxt][wrow * 32 + wcol] = *(const int4*)gw;
        }
        v16h af[2], bf[2];
#pragma unroll
        for (int i = 0; i < 2; ++i) {
            af[i] = *(const v16h*)&lsA[cur][(wm * 32 + i * 16 + l16) * 32 + l2 * 16];
            bf[i] = *(const v16h*)&lsW[cur][(wn * 32 + i * 16 + l16) * 32 + l2 * 16];
        }
#pragma unroll
        for (int i = 0; i < 2; ++i)
#pragma unroll
            for (int j = 0; j < 2; ++j)
                c[i][j] = __builtin_amdgcn_wmma_f32_16x16x32_f16(
                    false, af[i], false, bf[j], (short)0, c[i][j], false, false);
        __syncthreads();
    }

    // epilogue: element (r,lane) -> m = r + 8*l2, n = l16 within 16x16 tile
#pragma unroll
    for (int i = 0; i < 2; ++i) {
#pragma unroll
        for (int j = 0; j < 2; ++j) {
            const int n = nblk + wn * 32 + j * 16 + l16;
            float bv = HAS_BIAS ? bias[n] : 0.f;
#pragma unroll
            for (int r = 0; r < 8; ++r) {
                const int m = mblk + wm * 32 + i * 16 + r + l2 * 8;
                float v = c[i][j][r] + bv;
                if (RESIDUAL) v += res[(size_t)m * N + n];
                if (GELU_OUT) v = 0.5f * v * (1.f + erff(v * 0.70710678118f));
                if (OUT_F16) outH[(size_t)m * N + n] = (half_t)v;
                else         outF[(size_t)m * N + n] = v;
            }
        }
    }
}

// ---------------------------------------------------------------------------
// Flash attention: block = 4 waves; each wave owns a 16-row Q tile; 32 keys
// per iteration (two 16x16 S tiles -> online softmax -> P@V via LDS-transposed
// P fragment). Vt is pre-transposed [B*H, DH, NK] so PV B-frags are contiguous.
// ---------------------------------------------------------------------------
template<bool HAS_BIAS>
__global__ __launch_bounds__(128)
void flash_attn_kernel(const half_t* __restrict__ Qb, const half_t* __restrict__ Kb,
                       const half_t* __restrict__ Vt, half_t* __restrict__ O,
                       int qStride, int kStride,
                       const float* __restrict__ sims, const float* __restrict__ meansim,
                       const unsigned char* __restrict__ mask) {
    __shared__ __align__(32) half_t pShared[4][16 * 32];
    const int wave = threadIdx.x >> 5, lane = threadIdx.x & 31;
    const int l2 = lane >> 4, l16 = lane & 15;
    const int bh = blockIdx.y, b = bh / HH, h = bh - b * HH;
    const int q0 = blockIdx.x * 64 + wave * 16;

    const half_t* Q  = Qb + (size_t)b * NQ  * qStride + h * DHH;
    const half_t* Kp = Kb + (size_t)b * NKK * kStride + h * DHH;
    const half_t* Vp = Vt + (size_t)bh * DHH * NKK;

    v16h qf[2];
#pragma unroll
    for (int d = 0; d < 2; ++d)
        qf[d] = *(const v16h*)&Q[(size_t)(q0 + l16) * qStride + d * 32 + l2 * 16];

    v8f o[4] = {};
    float mrow[8], lrow[8];
#pragma unroll
    for (int r = 0; r < 8; ++r) { mrow[r] = -1e30f; lrow[r] = 0.f; }

    for (int kb = 0; kb < NKK; kb += 32) {
        v8f s[2];
#pragma unroll
        for (int j = 0; j < 2; ++j) {
            v8f acc = {};
#pragma unroll
            for (int d = 0; d < 2; ++d) {
                v16h kf = *(const v16h*)&Kp[(size_t)(kb + j * 16 + l16) * kStride + d * 32 + l2 * 16];
                acc = __builtin_amdgcn_wmma_f32_16x16x32_f16(
                    false, qf[d], false, kf, (short)0, acc, false, false);
            }
            s[j] = acc;
        }
        // scale + (bias,mask)
#pragma unroll
        for (int j = 0; j < 2; ++j) {
            const int kcol = kb + j * 16 + l16;
#pragma unroll
            for (int r = 0; r < 8; ++r) {
                const int qrow = q0 + r + l2 * 8;
                float v = s[j][r] * kScale;
                if (HAS_BIAS) {
                    v += sims[((size_t)bh * NQ + qrow) * NKK + kcol]
                       - meansim[((size_t)b * NQ + qrow) * NKK + kcol];
                    if (!mask[(size_t)qrow * NKK + kcol]) v = -1e30f;
                }
                s[j][r] = v;
            }
        }
        // online softmax: row values live across the 16-lane half (xor 1,2,4,8)
#pragma unroll
        for (int r = 0; r < 8; ++r) {
            float v = fmaxf(s[0][r], s[1][r]);
            v = fmaxf(v, __shfl_xor(v, 1)); v = fmaxf(v, __shfl_xor(v, 2));
            v = fmaxf(v, __shfl_xor(v, 4)); v = fmaxf(v, __shfl_xor(v, 8));
            float mnew  = fmaxf(mrow[r], v);
            float alpha = __expf(mrow[r] - mnew);
            mrow[r] = mnew;
            float e0 = __expf(s[0][r] - mnew);
            float e1 = __expf(s[1][r] - mnew);
            s[0][r] = e0; s[1][r] = e1;
            float rs = e0 + e1;
            rs += __shfl_xor(rs, 1); rs += __shfl_xor(rs, 2);
            rs += __shfl_xor(rs, 4); rs += __shfl_xor(rs, 8);
            lrow[r] = lrow[r] * alpha + rs;
#pragma unroll
            for (int j2 = 0; j2 < 4; ++j2) o[j2][r] *= alpha;
        }
        // P -> LDS (S layout) -> A-fragment layout (wave-local, no barrier)
        half_t* ps = pShared[wave];
#pragma unroll
        for (int j = 0; j < 2; ++j)
#pragma unroll
            for (int r = 0; r < 8; ++r)
                ps[(r + l2 * 8) * 32 + j * 16 + l16] = (half_t)s[j][r];
        asm volatile("s_wait_dscnt 0" ::: "memory");
        v16h pa = *(const v16h*)&ps[l16 * 32 + l2 * 16];
        // O += P @ V
#pragma unroll
        for (int j2 = 0; j2 < 4; ++j2) {
            v16h vf = *(const v16h*)&Vp[(size_t)(j2 * 16 + l16) * NKK + kb + l2 * 16];
            o[j2] = __builtin_amdgcn_wmma_f32_16x16x32_f16(
                false, pa, false, vf, (short)0, o[j2], false, false);
        }
    }
    // normalize + store to [B,NQ,H,DH] (== [B*NQ, C])
#pragma unroll
    for (int r = 0; r < 8; ++r) {
        const float inv = 1.f / fmaxf(lrow[r], 1e-30f);
        const int qrow = q0 + r + l2 * 8;
#pragma unroll
        for (int j2 = 0; j2 < 4; ++j2)
            O[((size_t)b * NQ + qrow) * CC + h * DHH + j2 * 16 + l16] = (half_t)(o[j2][r] * inv);
    }
}

// ---------------------------------------------------------------------------
// host launch
// ---------------------------------------------------------------------------
extern "C" void kernel_launch(void* const* d_in, const int* in_sizes, int n_in,
                              void* d_out, int out_size, void* d_ws, size_t ws_size,
                              hipStream_t stream) {
    (void)in_sizes; (void)n_in; (void)out_size; (void)ws_size;
    const float* x      = (const float*)d_in[0];
    const float* y      = (const float*)d_in[1];
    const unsigned char* mask = (const unsigned char*)d_in[4];
    const float* sims   = (const float*)d_in[5];
    const float* ln1_g = (const float*)d_in[6],  *ln1_b = (const float*)d_in[7];
    const float* ln2_g = (const float*)d_in[8],  *ln2_b = (const float*)d_in[9];
    const float* ln3_g = (const float*)d_in[10], *ln3_b = (const float*)d_in[11];
    const float* lny_g = (const float*)d_in[12], *lny_b = (const float*)d_in[13];
    const float* qkv_w  = (const float*)d_in[14];
    const float* aproj_w= (const float*)d_in[15], *aproj_b = (const float*)d_in[16];
    const float* pq_w   = (const float*)d_in[17], *pk_w = (const float*)d_in[18],
               * pv_w   = (const float*)d_in[19];
    const float* cproj_w= (const float*)d_in[20], *cproj_b = (const float*)d_in[21];
    const float* fc1_w  = (const float*)d_in[22], *fc1_b = (const float*)d_in[23];
    const float* fc2_w  = (const float*)d_in[24], *fc2_b = (const float*)d_in[25];

    const size_t S1 = (size_t)MM * CC;                 // 3.1M elems
    char* p = (char*)d_ws;
    auto alloc = [&](size_t bytes) { char* r = p; p += (bytes + 255) & ~(size_t)255; return r; };

    half_t* w_qkv  = (half_t*)alloc((size_t)3*CC*CC*2);
    half_t* w_ap   = (half_t*)alloc((size_t)CC*CC*2);
    half_t* w_pq   = (half_t*)alloc((size_t)CC*CC*2);
    half_t* w_pk   = (half_t*)alloc((size_t)CC*CC*2);
    half_t* w_pv   = (half_t*)alloc((size_t)CC*CC*2);
    half_t* w_cp   = (half_t*)alloc((size_t)CC*CC*2);
    half_t* w_fc1  = (half_t*)alloc((size_t)4*CC*CC*2);
    half_t* w_fc2  = (half_t*)alloc((size_t)4*CC*CC*2);
    half_t* qkv16  = (half_t*)alloc((size_t)MM*3*CC*2);   // later reused: cq|ck|cv
    half_t* a16    = (half_t*)alloc(S1*2);
    half_t* b16    = (half_t*)alloc(S1*2);
    half_t* vt16   = (half_t*)alloc(S1*2);
    half_t* ao16   = (half_t*)alloc(S1*2);
    float * X      = (float*)alloc(S1*4);
    float * X2     = (float*)alloc(S1*4);
    float * msim   = (float*)alloc((size_t)BB*NQ*NKK*4);
    half_t* h16    = (half_t*)alloc((size_t)MM*4*CC*2);

    auto cvt = [&](const float* src, half_t* dst, int n) {
        cvt_f16_kernel<<<(n + 255) / 256, 256, 0, stream>>>(src, dst, n);
    };
    cvt(qkv_w,  w_qkv, 3*CC*CC); cvt(aproj_w, w_ap, CC*CC);
    cvt(pq_w,   w_pq,  CC*CC);   cvt(pk_w,    w_pk, CC*CC);
    cvt(pv_w,   w_pv,  CC*CC);   cvt(cproj_w, w_cp, CC*CC);
    cvt(fc1_w,  w_fc1, 4*CC*CC); cvt(fc2_w,   w_fc2, 4*CC*CC);

    head_mean_kernel<<<(BB*NQ*NKK + 255) / 256, 256, 0, stream>>>(sims, msim);

    // ---- self attention ----
    layernorm_f16_kernel<<<MM, 256, 0, stream>>>(x, ln1_g, ln1_b, a16);
    gemm_wmma_kernel<false,false,false,true><<<dim3(MM/128, (3*CC)/64), 256, 0, stream>>>(
        a16, w_qkv, nullptr, nullptr, qkv16, nullptr, MM, 3*CC, CC);
    transpose_v_kernel<<<dim3(NKK/64, DHH, BB*HH), 64, 0, stream>>>(qkv16 + 2*CC, vt16, 3*CC);
    flash_attn_kernel<false><<<dim3(NQ/64, BB*HH), 128, 0, stream>>>(
        qkv16, qkv16 + CC, vt16, ao16, 3*CC, 3*CC, nullptr, nullptr, nullptr);
    gemm_wmma_kernel<true,true,false,false><<<dim3(MM/128, CC/64), 256, 0, stream>>>(
        ao16, w_ap, aproj_b, x, nullptr, X, MM, CC, CC);

    // ---- cross attention ----
    half_t* cq16 = qkv16;            // reuse qkv buffer
    half_t* ck16 = qkv16 + S1;
    half_t* cv16 = qkv16 + 2*S1;
    layernorm_f16_kernel<<<MM, 256, 0, stream>>>(y, lny_g, lny_b, b16);
    layernorm_f16_kernel<<<MM, 256, 0, stream>>>(X, ln2_g, ln2_b, a16);
    gemm_wmma_kernel<false,false,false,true><<<dim3(MM/128, CC/64), 256, 0, stream>>>(
        a16, w_pq, nullptr, nullptr, cq16, nullptr, MM, CC, CC);
    gemm_wmma_kernel<false,false,false,true><<<dim3(MM/128, CC/64), 256, 0, stream>>>(
        b16, w_pk, nullptr, nullptr, ck16, nullptr, MM, CC, CC);
    gemm_wmma_kernel<false,false,false,true><<<dim3(MM/128, CC/64), 256, 0, stream>>>(
        b16, w_pv, nullptr, nullptr, cv16, nullptr, MM, CC, CC);
    transpose_v_kernel<<<dim3(NKK/64, DHH, BB*HH), 64, 0, stream>>>(cv16, vt16, CC);
    flash_attn_kernel<true><<<dim3(NQ/64, BB*HH), 128, 0, stream>>>(
        cq16, ck16, vt16, ao16, CC, CC, sims, msim, mask);
    gemm_wmma_kernel<true,true,false,false><<<dim3(MM/128, CC/64), 256, 0, stream>>>(
        ao16, w_cp, cproj_b, X, nullptr, X2, MM, CC, CC);

    // ---- MLP ----
    layernorm_f16_kernel<<<MM, 256, 0, stream>>>(X2, ln3_g, ln3_b, a16);
    gemm_wmma_kernel<true,false,true,true><<<dim3(MM/128, (4*CC)/64), 256, 0, stream>>>(
        a16, w_fc1, fc1_b, nullptr, h16, nullptr, MM, 4*CC, CC);
    gemm_wmma_kernel<true,true,false,false><<<dim3(MM/128, CC/64), 256, 0, stream>>>(
        h16, w_fc2, fc2_b, X2, nullptr, (float*)d_out, MM, CC, 4*CC);

    // ---- y passthrough (second tuple element) ----
    hipMemcpyAsync((float*)d_out + S1, y, S1 * sizeof(float),
                   hipMemcpyDeviceToDevice, stream);
}
